// GATEncoder_v2_29557964931455
// MI455X (gfx1250) — matrix operand order repaired and checked by reading, not compiled, not run on hardware
//
#include <hip/hip_runtime.h>

// ---------------------------------------------------------------------------
// GATv2 encoder (4 layers, H=4 heads) for gfx1250 (MI455X).
// Dense transforms: LDS-staged bf16 WMMA (v_wmma_f32_16x16x32_bf16,
// ds_load_b128 fragment feeds). Segment softmax/aggregation: L2 atomics.
// ---------------------------------------------------------------------------

constexpr int Nn  = 30000;           // nodes
constexpr int Ee  = 480000;          // raw edges
constexpr int Ep  = Ee + Nn;         // + self loops
constexpr int HH  = 4;               // heads
constexpr int D1  = 128;             // H * fo (all layers emit 128 wide)
constexpr int FO  = 32;              // per-head features
constexpr float SLOPE = 0.2f;

typedef __bf16  v16bf __attribute__((ext_vector_type(16)));
typedef float   v8f   __attribute__((ext_vector_type(8)));

union Frag {
    v16bf v;
    unsigned short u[16];
    uint4 q[2];
};

__device__ inline unsigned short f2bf(float f) {
    unsigned u = __float_as_uint(f);
    u += 0x7FFFu + ((u >> 16) & 1u);        // round-to-nearest-even
    return (unsigned short)(u >> 16);
}

// float -> monotonically ordered unsigned key (for atomicMax segment-max)
__device__ inline unsigned fkey(float f) {
    unsigned b = __float_as_uint(f);
    return (b & 0x80000000u) ? ~b : (b | 0x80000000u);
}
__device__ inline float fkey_inv(unsigned k) {
    unsigned b = (k & 0x80000000u) ? (k & 0x7FFFFFFFu) : ~k;
    return __uint_as_float(b);
}

// ---------------------------------------------------------------------------
// WMMA GEMM: D[N,128] = A[N,K] @ W[K,128] (+bias) [+residual, relu]
// MODE 0: D = A@W + bias            (xl / xr transforms)
// MODE 1: D = relu(A@W + bias + R)  (residual blocks)
// Block = 256 threads = 8 waves. Block covers 16 rows x 128 cols; wave w
// owns column tile w. A-strip and transposed W staged in LDS as bf16 once,
// hot loop is ds_load_b128 + v_wmma only. Row stride of A == K.
// ---------------------------------------------------------------------------
template <int K, int MODE>
__global__ __launch_bounds__(256)
void wmma_gemm_lds(const float* __restrict__ A, const float* __restrict__ W,
                   const float* __restrict__ bias, const float* __restrict__ R,
                   float* __restrict__ D) {
    constexpr int KP = K + 8;                         // pad: 16B stays aligned,
                                                      // banks spread 4/col
    __shared__ unsigned short As[16 * KP];            // A strip, bf16
    __shared__ unsigned short WT[D1 * KP];            // W transposed, bf16

    const int tid  = threadIdx.x;
    const int lane = tid & 31;
    const int wave = tid >> 5;                        // 0..7 -> column tile
    const int l15  = lane & 15;
    const int lh   = lane >> 4;                       // lane-half select
    const int row0 = blockIdx.x * 16;
    const int col  = wave * 16 + l15;

    // --- stage A strip (16 rows x K) as bf16, packed uint2 stores ---------
    for (int q = tid; q < 16 * K / 4; q += 256) {
        int r  = q / (K / 4);
        int k4 = (q % (K / 4)) * 4;
        float4 v = *(const float4*)(A + (size_t)(row0 + r) * K + k4);
        unsigned p0 = (unsigned)f2bf(v.x) | ((unsigned)f2bf(v.y) << 16);
        unsigned p1 = (unsigned)f2bf(v.z) | ((unsigned)f2bf(v.w) << 16);
        *(uint2*)(As + r * KP + k4) = make_uint2(p0, p1);
    }
    // --- stage W transposed (WT[c][k]) as bf16 ----------------------------
    for (int q = tid; q < K * D1 / 4; q += 256) {
        int k  = q / (D1 / 4);
        int c4 = (q % (D1 / 4)) * 4;
        float4 v = *(const float4*)(W + (size_t)k * D1 + c4);
        WT[(c4 + 0) * KP + k] = f2bf(v.x);
        WT[(c4 + 1) * KP + k] = f2bf(v.y);
        WT[(c4 + 2) * KP + k] = f2bf(v.z);
        WT[(c4 + 3) * KP + k] = f2bf(v.w);
    }
    __syncthreads();

    // --- main loop: pure ds_load_b128 + wmma ------------------------------
    v8f c = {};
#pragma unroll
    for (int kk = 0; kk < K; kk += 32) {
        Frag a, b;
        // A frag: lane row = l15; K = {kk+8*lh+j} U {kk+16+8*lh+j}, j=0..7
        const unsigned short* ap = As + l15 * KP + kk + lh * 8;
        a.q[0] = *(const uint4*)(ap);
        a.q[1] = *(const uint4*)(ap + 16);
        // B frag: lane col; K = {kk+8*lh+j} U {kk+16+8*lh+j}, j=0..7
        const unsigned short* bp = WT + col * KP + kk + lh * 8;
        b.q[0] = *(const uint4*)(bp);
        b.q[1] = *(const uint4*)(bp + 16);
        c = __builtin_amdgcn_wmma_f32_16x16x32_bf16(
                false, a.v, false, b.v, (short)0, c, false, false);
    }

    const float bb = bias[col];
#pragma unroll
    for (int i = 0; i < 8; ++i) {
        int m = row0 + i + (lh << 3);                 // C/D layout: M = i + 8*lh
        size_t idx = (size_t)m * D1 + col;
        float v = c[i] + bb;
        if (MODE == 1) v = fmaxf(v + R[idx], 0.0f);
        D[idx] = v;
    }
}

// ---------------------------------------------------------------------------
__global__ void init_buffers(float* __restrict__ acc, float* __restrict__ dn,
                             unsigned* __restrict__ mk) {
    int i = blockIdx.x * blockDim.x + threadIdx.x;
    if (i < Nn * D1) acc[i] = 0.0f;
    if (i < Nn * HH) { dn[i] = 0.0f; mk[i] = 0u; }   // key floor == most-negative
}

// e_ij = att_h . LeakyReLU(xl[src] + xr[dst]); segment-max via atomicMax key
__global__ void edge_scores(const int* __restrict__ ei,
                            const float* __restrict__ xl,
                            const float* __restrict__ xr,
                            const float* __restrict__ att,
                            float* __restrict__ eb, unsigned* __restrict__ mk) {
    int idx = blockIdx.x * blockDim.x + threadIdx.x;
    if (idx >= Ep * HH) return;
    int e = idx >> 2, h = idx & 3;
    int src = (e < Ee) ? ei[e]      : (e - Ee);
    int dst = (e < Ee) ? ei[Ee + e] : (e - Ee);

    const float4* pl = (const float4*)(xl + (size_t)src * D1 + h * FO);
    const float4* pr = (const float4*)(xr + (size_t)dst * D1 + h * FO);
    const float4* pa = (const float4*)(att + h * FO);
    float s = 0.0f;
#pragma unroll
    for (int q = 0; q < FO / 4; ++q) {
        float4 a = pl[q], b = pr[q], w = pa[q];
        float v0 = a.x + b.x, v1 = a.y + b.y, v2 = a.z + b.z, v3 = a.w + b.w;
        v0 = v0 > 0.f ? v0 : SLOPE * v0;  v1 = v1 > 0.f ? v1 : SLOPE * v1;
        v2 = v2 > 0.f ? v2 : SLOPE * v2;  v3 = v3 > 0.f ? v3 : SLOPE * v3;
        s += v0 * w.x + v1 * w.y + v2 * w.z + v3 * w.w;
    }
    eb[idx] = s;
    atomicMax(&mk[dst * HH + h], fkey(s));
}

// a = exp(e - max); denom[dst] += a; acc[dst] += a * xl[src]
__global__ void edge_accum(const int* __restrict__ ei,
                           const float* __restrict__ xl,
                           const float* __restrict__ eb,
                           const unsigned* __restrict__ mk,
                           float* __restrict__ dn, float* __restrict__ acc) {
    int idx = blockIdx.x * blockDim.x + threadIdx.x;
    if (idx >= Ep * HH) return;
    int e = idx >> 2, h = idx & 3;
    int src = (e < Ee) ? ei[e]      : (e - Ee);
    int dst = (e < Ee) ? ei[Ee + e] : (e - Ee);

    float m = fkey_inv(mk[dst * HH + h]);
    float a = __expf(eb[idx] - m);
    atomicAdd(&dn[dst * HH + h], a);

    const float* pl = xl + (size_t)src * D1 + h * FO;
    float* pacc     = acc + (size_t)dst * D1 + h * FO;
#pragma unroll
    for (int f = 0; f < FO; ++f) atomicAdd(&pacc[f], a * pl[f]);
}

// concat layers: out = relu(acc/denom + bo)   [N,128]
__global__ void fin_concat(const float* __restrict__ acc,
                           const float* __restrict__ dn,
                           const float* __restrict__ bo, float* __restrict__ out) {
    int idx = blockIdx.x * blockDim.x + threadIdx.x;
    if (idx >= Nn * D1) return;
    int n = idx >> 7, c = idx & 127, h = c >> 5;
    float v = acc[idx] / (dn[n * HH + h] + 1e-16f) + bo[c];
    out[idx] = fmaxf(v, 0.0f);
}

// last layer: out = relu(mean_h(acc/denom) + bo)   [N,32]
__global__ void fin_mean(const float* __restrict__ acc,
                         const float* __restrict__ dn,
                         const float* __restrict__ bo, float* __restrict__ out) {
    int idx = blockIdx.x * blockDim.x + threadIdx.x;
    if (idx >= Nn * FO) return;
    int n = idx >> 5, f = idx & 31;
    float s = 0.0f;
#pragma unroll
    for (int h = 0; h < HH; ++h)
        s += acc[(size_t)n * D1 + h * FO + f] / (dn[n * HH + h] + 1e-16f);
    out[idx] = fmaxf(s * 0.25f + bo[f], 0.0f);
}

// ---------------------------------------------------------------------------
extern "C" void kernel_launch(void* const* d_in, const int* in_sizes, int n_in,
                              void* d_out, int out_size, void* d_ws, size_t ws_size,
                              hipStream_t stream) {
    (void)in_sizes; (void)n_in; (void)out_size; (void)ws_size;

    const float* x  = (const float*)d_in[0];
    const int*   ei = (const int*)d_in[1];
    // d_in[2] = frame_mask (unused by reference math)
    const float *Wl[4], *Wr[4], *bl[4], *br[4], *att[4], *bo[4];
    for (int l = 0; l < 4; ++l) {
        int b = 3 + l * 6;
        Wl[l]  = (const float*)d_in[b + 0];
        Wr[l]  = (const float*)d_in[b + 1];
        bl[l]  = (const float*)d_in[b + 2];
        br[l]  = (const float*)d_in[b + 3];
        att[l] = (const float*)d_in[b + 4];
        bo[l]  = (const float*)d_in[b + 5];
    }
    const float* res1_W = (const float*)d_in[27];
    const float* res1_b = (const float*)d_in[28];
    const float* res2_W = (const float*)d_in[29];
    const float* res2_b = (const float*)d_in[30];

    const size_t NF = (size_t)Nn * D1;
    float* XL  = (float*)d_ws;
    float* XR  = XL  + NF;
    float* ACC = XR  + NF;
    float* X1  = ACC + NF;
    float* X2  = X1  + NF;
    float* X3  = X2  + NF;
    float* EB  = X3  + NF;
    float* DN  = EB  + (size_t)Ep * HH;
    unsigned* MK = (unsigned*)(DN + (size_t)Nn * HH);

    const int gemmGrid = Nn / 16;                       // 1875 (exact)
    const int eTh = Ep * HH;
    const int eGrid = (eTh + 255) / 256;
    const int iGrid = (Nn * D1 + 255) / 256;

    auto run_edges = [&](int l) {
        init_buffers<<<iGrid, 256, 0, stream>>>(ACC, DN, MK);
        edge_scores<<<eGrid, 256, 0, stream>>>(ei, XL, XR, att[l], EB, MK);
        edge_accum <<<eGrid, 256, 0, stream>>>(ei, XL, EB, MK, DN, ACC);
    };

    // Layer 1: x[N,64] -> X1
    wmma_gemm_lds<64, 0><<<gemmGrid, 256, 0, stream>>>(x, Wl[0], bl[0], nullptr, XL);
    wmma_gemm_lds<64, 0><<<gemmGrid, 256, 0, stream>>>(x, Wr[0], br[0], nullptr, XR);
    run_edges(0);
    fin_concat<<<iGrid, 256, 0, stream>>>(ACC, DN, bo[0], X1);

    // Layer 2 + residual 1: X2 = relu(h @ res1_W + res1_b + X1)
    wmma_gemm_lds<128, 0><<<gemmGrid, 256, 0, stream>>>(X1, Wl[1], bl[1], nullptr, XL);
    wmma_gemm_lds<128, 0><<<gemmGrid, 256, 0, stream>>>(X1, Wr[1], br[1], nullptr, XR);
    run_edges(1);
    fin_concat<<<iGrid, 256, 0, stream>>>(ACC, DN, bo[1], XL);     // h -> XL
    wmma_gemm_lds<128, 1><<<gemmGrid, 256, 0, stream>>>(XL, res1_W, res1_b, X1, X2);

    // Layer 3 + residual 2: X3 = relu(h @ res2_W + res2_b + X2)
    wmma_gemm_lds<128, 0><<<gemmGrid, 256, 0, stream>>>(X2, Wl[2], bl[2], nullptr, XL);
    wmma_gemm_lds<128, 0><<<gemmGrid, 256, 0, stream>>>(X2, Wr[2], br[2], nullptr, XR);
    run_edges(2);
    fin_concat<<<iGrid, 256, 0, stream>>>(ACC, DN, bo[2], XL);     // h -> XL
    wmma_gemm_lds<128, 1><<<gemmGrid, 256, 0, stream>>>(XL, res2_W, res2_b, X2, X3);

    // Layer 4 (mean over heads) -> d_out [N,32]
    wmma_gemm_lds<128, 0><<<gemmGrid, 256, 0, stream>>>(X3, Wl[3], bl[3], nullptr, XL);
    wmma_gemm_lds<128, 0><<<gemmGrid, 256, 0, stream>>>(X3, Wr[3], br[3], nullptr, XR);
    run_edges(3);
    const int mGrid = (Nn * FO + 255) / 256;
    fin_mean<<<mGrid, 256, 0, stream>>>(ACC, DN, bo[3], (float*)d_out);
}
